// GuidanceNodeGenerator_58952721105294
// MI455X (gfx1250) — compile-verified, matrix operand order checked by reading, not run on hardware
//
#include <hip/hip_runtime.h>

#define HD 128  // feature dim (D == H == 128)

typedef __attribute__((ext_vector_type(16))) __bf16 v16bf;
typedef __attribute__((ext_vector_type(8)))  __bf16 v8bf;
typedef __attribute__((ext_vector_type(8)))  float  v8f;

__device__ __forceinline__ unsigned short f2bf(float f) {
  union { float f; unsigned u; } v; v.f = f;
  unsigned r = v.u + 0x7FFFu + ((v.u >> 16) & 1u);
  return (unsigned short)(r >> 16);
}

// ---------------- utility ----------------
__global__ void k_zero(float* __restrict__ p, long n) {
  long i = (long)blockIdx.x * blockDim.x + threadIdx.x;
  if (i < n) p[i] = 0.f;
}

__global__ void k_count_deg(const int* __restrict__ dst, float* __restrict__ deg, int E) {
  int e = blockIdx.x * blockDim.x + threadIdx.x;
  if (e < E) atomicAdd(&deg[dst[e]], 1.f);
}

__global__ void k_dinv(float* __restrict__ deg, int n) {
  int i = blockIdx.x * blockDim.x + threadIdx.x;
  if (i < n) deg[i] = rsqrtf(deg[i] + 1.f);  // +1 for self loop, always > 0
}

__global__ void k_f32_to_bf16(const float* __restrict__ in, unsigned short* __restrict__ out, long n) {
  long i = (long)blockIdx.x * blockDim.x + threadIdx.x;
  if (i < n) out[i] = f2bf(in[i]);
}

// Wt[n][k] = W[k][n], bf16. grid 128, block 128.
__global__ void k_transpose_w(const float* __restrict__ W, unsigned short* __restrict__ Wt) {
  int k = blockIdx.x, n = threadIdx.x;
  Wt[n * HD + k] = f2bf(W[k * HD + n]);
}

// ---------------- WMMA GEMM: C[M x128] = A_bf16[M x128] @ W, with Wt[n][k] staged ----------------
// One wave per 16x128 output strip: A fragment loaded ONCE per K-step, reused across all 8
// column tiles (8 accumulators = 64 VGPRs). K = 128 -> 4 K-steps x 8 tiles = 32 WMMAs/wave.
__global__ __launch_bounds__(32) void k_gemm_wmma(const unsigned short* __restrict__ Abf,
                                                  const unsigned short* __restrict__ Wtbf,
                                                  float* __restrict__ C) {
  const int tm = blockIdx.x;        // row tile (16 rows)
  const int lane = threadIdx.x;     // 0..31 (wave32)
  const int half = lane >> 4;       // 0 -> K base 0, 1 -> K base 8 (ISA 16-bit A/B layout)
  const int l15  = lane & 15;
  const __bf16* A  = (const __bf16*)Abf;
  const __bf16* Bt = (const __bf16*)Wtbf;
  const long arow = (long)tm * 16 + l15;  // A row held by this lane

  v8f acc[8];
#pragma unroll
  for (int tn = 0; tn < 8; ++tn) acc[tn] = (v8f){};

#pragma unroll
  for (int kk = 0; kk < 4; ++kk) {
    const int kb = kk * 32 + half * 8;
    // A fragment: lanes 0-15 hold K {kb..kb+7, kb+16..kb+23} of row l15 (half=0),
    // lanes 16-31 the +8 K-offset — two contiguous 16B loads.
    v8bf a0 = *(const v8bf*)(A + arow * HD + kb);
    v8bf a1 = *(const v8bf*)(A + arow * HD + kb + 16);
    v16bf av = __builtin_shufflevector(a0, a1, 0,1,2,3,4,5,6,7,8,9,10,11,12,13,14,15);
#pragma unroll
    for (int tn = 0; tn < 8; ++tn) {
      const long bcol = (long)tn * 16 + l15;  // W column held by this lane (Wt row)
      v8bf b0 = *(const v8bf*)(Bt + bcol * HD + kb);
      v8bf b1 = *(const v8bf*)(Bt + bcol * HD + kb + 16);
      v16bf bv = __builtin_shufflevector(b0, b1, 0,1,2,3,4,5,6,7,8,9,10,11,12,13,14,15);
      acc[tn] = __builtin_amdgcn_wmma_f32_16x16x32_bf16(false, av, false, bv,
                                                        (short)0, acc[tn], false, false);
    }
  }
  // C/D layout: VGPR r -> row 16*tm + half*8 + r, col 16*tn + (lane&15)
#pragma unroll
  for (int tn = 0; tn < 8; ++tn) {
    float* Cp = C + ((long)tm * 16 + half * 8) * HD + tn * 16 + l15;
#pragma unroll
    for (int r = 0; r < 8; ++r) Cp[(long)r * HD] = acc[tn][r];
  }
}

// ---------------- edge scatter-add (one wave per edge, float4 per lane) ----------------
__global__ void k_aggregate(const int* __restrict__ src, const int* __restrict__ dst,
                            const float* __restrict__ dinv, const float* __restrict__ t,
                            float* __restrict__ agg, int E) {
  int e = blockIdx.x * (blockDim.x >> 5) + (threadIdx.x >> 5);
  if (e >= E) return;
  int lane = threadIdx.x & 31;
  int s = src[e], d = dst[e];
  float nrm = dinv[s] * dinv[d];
  float4 v = ((const float4*)(t + (long)s * HD))[lane];
  float* ap = agg + (long)d * HD + lane * 4;
  atomicAdd(ap + 0, v.x * nrm);
  atomicAdd(ap + 1, v.y * nrm);
  atomicAdd(ap + 2, v.z * nrm);
  atomicAdd(ap + 3, v.w * nrm);
}

// out = relu(agg + t*dinv^2 + b); bf16 output (feeds next GEMM)
__global__ void k_finalize_bf(const float* __restrict__ t, const float* __restrict__ agg,
                              const float* __restrict__ dinv, const float* __restrict__ b,
                              unsigned short* __restrict__ out, long n) {
  long i = (long)blockIdx.x * blockDim.x + threadIdx.x;
  if (i >= n) return;
  int node = (int)(i >> 7), f = (int)(i & 127);
  float di = dinv[node];
  float v = agg[i] + t[i] * di * di + b[f];
  out[i] = f2bf(fmaxf(v, 0.f));
}

// fp32 output (last layer)
__global__ void k_finalize_f32(const float* __restrict__ t, const float* __restrict__ agg,
                               const float* __restrict__ dinv, const float* __restrict__ b,
                               float* __restrict__ out, long n) {
  long i = (long)blockIdx.x * blockDim.x + threadIdx.x;
  if (i >= n) return;
  int node = (int)(i >> 7), f = (int)(i & 127);
  float di = dinv[node];
  float v = agg[i] + t[i] * di * di + b[f];
  out[i] = fmaxf(v, 0.f);
}

// zero rows of bf16 x for batch nodes (env view). grid B, block 128.
__global__ void k_mask_rows(unsigned short* __restrict__ xbf, const int* __restrict__ bn) {
  xbf[(long)bn[blockIdx.x] * HD + threadIdx.x] = 0;
}

// gather gf -> d_out, y, prototype accumulation. grid B, block 128.
__global__ void k_gather_gf(const float* __restrict__ h2, const int* __restrict__ bn,
                            const int* __restrict__ labels, float* __restrict__ gf,
                            float* __restrict__ y, float* __restrict__ proto,
                            float* __restrict__ pos_cnt) {
  int j = blockIdx.x, f = threadIdx.x;
  int node = bn[j];
  float v = h2[(long)node * HD + f];
  gf[(long)j * HD + f] = v;
  float yy = (labels[node] == 1) ? 1.f : 0.f;
  if (f == 0) {
    y[j] = yy;
    if (yy > 0.f) atomicAdd(pos_cnt, 1.f);
  }
  if (yy > 0.f) atomicAdd(&proto[f], v);
}

__global__ void k_gather_env(const float* __restrict__ h2, const int* __restrict__ bn,
                             float* __restrict__ envf) {
  int j = blockIdx.x, f = threadIdx.x;
  envf[(long)j * HD + f] = h2[(long)bn[j] * HD + f];
}

// vctx = W_ctx @ (proto / pos_cnt). one block of 128.
__global__ void k_ctx_vec(const float* __restrict__ Wc, const float* __restrict__ proto,
                          const float* __restrict__ pos_cnt, float* __restrict__ vctx) {
  int d = threadIdx.x;
  float pc = *pos_cnt;
  float inv = (pc > 0.f) ? 1.f / pc : 0.f;
  float s = 0.f;
  for (int e = 0; e < HD; ++e) s += Wc[d * HD + e] * (proto[e] * inv);
  vctx[d] = s;
}

__device__ __forceinline__ float softplusf(float z) {
  return (z > 0.f) ? z + log1pf(expf(-z)) : log1pf(expf(z));
}

// ctx loss: z_j = gf_j . vctx + b_ctx ; accumulate softplus(z) - z*y
__global__ void k_ctx_loss(const float* __restrict__ gf, const float* __restrict__ vctx,
                           const float* __restrict__ y, const float* __restrict__ bc,
                           float* __restrict__ loss, int B) {
  int j = blockIdx.x * blockDim.x + threadIdx.x;
  if (j >= B) return;
  float z = bc[0];
  for (int e = 0; e < HD; ++e) z += gf[(long)j * HD + e] * vctx[e];
  atomicAdd(loss, softplusf(z) - z * y[j]);
}

// local loss: z_j = env_f_j^T W_loc gf_j + b_loc, target (1-y). block-per-node, 128 threads.
__global__ void k_loc_loss(const float* __restrict__ envf, const float* __restrict__ gf,
                           const float* __restrict__ Wl, const float* __restrict__ y,
                           const float* __restrict__ bl, float* __restrict__ loss) {
  __shared__ float gsh[HD];
  __shared__ float red[HD];
  int j = blockIdx.x, d = threadIdx.x;
  gsh[d] = gf[(long)j * HD + d];
  __syncthreads();
  float w = 0.f;
  for (int e = 0; e < HD; ++e) w += Wl[d * HD + e] * gsh[e];
  red[d] = w * envf[(long)j * HD + d];
  __syncthreads();
  for (int s = 64; s > 0; s >>= 1) {
    if (d < s) red[d] += red[d + s];
    __syncthreads();
  }
  if (d == 0) {
    float z = red[0] + bl[0];
    atomicAdd(loss, softplusf(z) - z * (1.f - y[j]));
  }
}

__global__ void k_final(const float* __restrict__ loss_ctx, const float* __restrict__ loss_loc,
                        const float* __restrict__ pos_cnt, float* __restrict__ out, int B) {
  float pc = *pos_cnt;
  float total = 0.5f * (*loss_ctx / (float)B) + 0.5f * (*loss_loc / (float)B);
  if (!((pc > 0.f) && ((float)B - pc > 0.f))) total = 0.f;
  out[0] = total;
}

// ---------------- host-side orchestration ----------------
extern "C" void kernel_launch(void* const* d_in, const int* in_sizes, int n_in,
                              void* d_out, int out_size, void* d_ws, size_t ws_size,
                              hipStream_t stream) {
  const float* x      = (const float*)d_in[0];
  const int*   ei     = (const int*)d_in[1];
  const int*   bn     = (const int*)d_in[2];
  const int*   labels = (const int*)d_in[3];
  const float* W1 = (const float*)d_in[4];
  const float* b1 = (const float*)d_in[5];
  const float* W2 = (const float*)d_in[6];
  const float* b2 = (const float*)d_in[7];
  const float* Wc = (const float*)d_in[8];
  const float* bc = (const float*)d_in[9];
  const float* Wl = (const float*)d_in[10];
  const float* bl = (const float*)d_in[11];

  const int N = in_sizes[0] / HD;   // 50000 (divisible by 16)
  const int E = in_sizes[1] / 2;    // 1.6M
  const int B = in_sizes[2];        // 4096
  const int* src = ei;
  const int* dst = ei + E;
  const long NH = (long)N * HD;

  // workspace carve-up
  char* ws = (char*)d_ws;
  size_t o = 0;
  auto alloc = [&](size_t bytes) { char* p = ws + o; o += (bytes + 255) & ~(size_t)255; return p; };
  float*          dinv = (float*)alloc((size_t)N * 4);
  float*          t    = (float*)alloc((size_t)NH * 4);
  float*          agg  = (float*)alloc((size_t)NH * 4);
  float*          h2   = (float*)alloc((size_t)NH * 4);
  unsigned short* bfA  = (unsigned short*)alloc((size_t)NH * 2);  // x (then masked x)
  unsigned short* bfB  = (unsigned short*)alloc((size_t)NH * 2);  // h1 / env h1
  unsigned short* wt1  = (unsigned short*)alloc((size_t)HD * HD * 2);
  unsigned short* wt2  = (unsigned short*)alloc((size_t)HD * HD * 2);
  float*          envf = (float*)alloc((size_t)B * HD * 4);
  float*          yv   = (float*)alloc((size_t)B * 4);
  float*          sml  = (float*)alloc(512 * 4);
  float* proto    = sml;         // [128]
  float* vctx     = sml + 128;   // [128]
  float* pos_cnt  = sml + 256;
  float* loss_ctx = sml + 257;
  float* loss_loc = sml + 258;
  float* gf = (float*)d_out;     // [B x H] written directly

  const int gemm_grid = N / 16;       // one wave per 16x128 strip
  const int agg_grid = (E + 7) / 8;   // 8 edges (waves) per 256-thread block
  const int nh_grid = (int)((NH + 255) / 256);

  // degrees -> dinv
  k_zero<<<(N + 255) / 256, 256, 0, stream>>>(dinv, N);
  k_count_deg<<<(E + 255) / 256, 256, 0, stream>>>(dst, dinv, E);
  k_dinv<<<(N + 255) / 256, 256, 0, stream>>>(dinv, N);

  // stage bf16 operands
  k_transpose_w<<<HD, HD, 0, stream>>>(W1, wt1);
  k_transpose_w<<<HD, HD, 0, stream>>>(W2, wt2);
  k_f32_to_bf16<<<nh_grid, 256, 0, stream>>>(x, bfA, NH);

  // ---- normal pass ----
  k_gemm_wmma<<<gemm_grid, 32, 0, stream>>>(bfA, wt1, t);
  k_zero<<<nh_grid, 256, 0, stream>>>(agg, NH);
  k_aggregate<<<agg_grid, 256, 0, stream>>>(src, dst, dinv, t, agg, E);
  k_finalize_bf<<<nh_grid, 256, 0, stream>>>(t, agg, dinv, b1, bfB, NH);

  k_gemm_wmma<<<gemm_grid, 32, 0, stream>>>(bfB, wt2, t);
  k_zero<<<nh_grid, 256, 0, stream>>>(agg, NH);
  k_aggregate<<<agg_grid, 256, 0, stream>>>(src, dst, dinv, t, agg, E);
  k_finalize_f32<<<nh_grid, 256, 0, stream>>>(t, agg, dinv, b2, h2, NH);

  // gather gf (into d_out) + prototype stats, before h2 is reused by env pass
  k_zero<<<2, 256, 0, stream>>>(sml, 512);
  k_gather_gf<<<B, HD, 0, stream>>>(h2, bn, labels, gf, yv, proto, pos_cnt);

  // ---- environment pass (mask batch rows of bf16 x in place) ----
  k_mask_rows<<<B, HD, 0, stream>>>(bfA, bn);
  k_gemm_wmma<<<gemm_grid, 32, 0, stream>>>(bfA, wt1, t);
  k_zero<<<nh_grid, 256, 0, stream>>>(agg, NH);
  k_aggregate<<<agg_grid, 256, 0, stream>>>(src, dst, dinv, t, agg, E);
  k_finalize_bf<<<nh_grid, 256, 0, stream>>>(t, agg, dinv, b1, bfB, NH);

  k_gemm_wmma<<<gemm_grid, 32, 0, stream>>>(bfB, wt2, t);
  k_zero<<<nh_grid, 256, 0, stream>>>(agg, NH);
  k_aggregate<<<agg_grid, 256, 0, stream>>>(src, dst, dinv, t, agg, E);
  k_finalize_f32<<<nh_grid, 256, 0, stream>>>(t, agg, dinv, b2, h2, NH);

  k_gather_env<<<B, HD, 0, stream>>>(h2, bn, envf);

  // ---- heads + losses ----
  k_ctx_vec<<<1, HD, 0, stream>>>(Wc, proto, pos_cnt, vctx);
  k_ctx_loss<<<(B + 255) / 256, 256, 0, stream>>>(gf, vctx, yv, bc, loss_ctx, B);
  k_loc_loss<<<B, HD, 0, stream>>>(envf, gf, Wl, yv, bl, loss_loc);
  k_final<<<1, 1, 0, stream>>>(loss_ctx, loss_loc, pos_cnt, gf + (long)B * HD, B);
}
